// AutoregressiveGeneratorConditioned_80298708566514
// MI455X (gfx1250) — matrix operand order repaired and checked by reading, not compiled, not run on hardware
//
#include <hip/hip_runtime.h>

// ---------------------------------------------------------------------------
// Types for CDNA5 WMMA (wave32): D(16x16 f32) = A(16x32 bf16) * B(32x16 bf16) + C
// ---------------------------------------------------------------------------
typedef __attribute__((ext_vector_type(16))) __bf16 v16bf;
typedef __attribute__((ext_vector_type(2)))  __bf16 bf16x2;
typedef __attribute__((ext_vector_type(8)))  float  v8f;

union Frag {
    unsigned int u[8];
    int4         q[2];
    v16bf        v;
};
union BfPair { __bf16 h[2]; unsigned int u; };

// pack two floats into a bf16x2 dword -> compiler emits v_cvt_pk_bf16_f32
static __device__ __forceinline__ unsigned int pack2_bf16(float a, float b) {
    BfPair p;
    p.h[0] = (__bf16)a;
    p.h[1] = (__bf16)b;
    return p.u;
}

static __device__ __forceinline__ float fsigmoid(float x) { return 1.f / (1.f + __expf(-x)); }
static __device__ __forceinline__ float ftanh(float x)    { return 2.f * fsigmoid(2.f * x) - 1.f; }

// ---------------------------------------------------------------------------
// Packed-weight workspace layout (dword offsets).  B fragments are stored
// lane-major: frag(1KB) = { lane0: v0..v7, lane1: v0..v7, ... } so each lane
// loads its 8 dwords with two global_load_b128.
// Fragment index for a KxN matrix: f = ntile*(K/32) + ktile.
// ---------------------------------------------------------------------------
constexpr int OFF_G1W = 0;        // gru1_W 128x384 : 96 frags
constexpr int OFF_G1U = 24576;    // gru1_U
constexpr int OFF_G2W = 49152;    // gru2_W
constexpr int OFF_G2U = 73728;    // gru2_U
constexpr int OFF_C0  = 98304;    // c0_W 128x64 : 16 frags
constexpr int OFF_E0  = 102400;   // e0_W
constexpr int OFF_R0  = 106496;   // r0_W
constexpr int OFF_Q0  = 110592;   // q0_W 128x32 : 8 frags
constexpr int OFF_C1  = 112640;   // c1_W 64x32  : 4 frags
constexpr int OFF_E1  = 113664;   // e1_W
constexpr int OFF_R1  = 114688;   // r1_W
constexpr int OFF_Q1  = 115712;   // q1_W 32x16  : 1 frag
constexpr int OFF_CONDF = 115968; // float: cond  1024x128
constexpr int OFF_H1F   = 247040; // float: h1
constexpr int OFF_H2F   = 378112; // float: h2
constexpr int OFF_CNT   = 509184; // int:   counts 1024

// LDS const-table offsets (floats; all multiples of 4 -> float4-aligned)
constexpr int G1B = 0,    G2B = 768;
constexpr int C0B = 1536, CLG = 1600, CLB = 1664, C1B = 1728, C2W = 1760, C2B = 1856;
constexpr int Q0B = 1860, Q1B = 1892, Q2W = 1908, Q2B = 1924;
constexpr int E0B = 1928, ELG = 1992, ELB = 2056, E1B = 2120, E2W = 2152, E2B = 2408;
constexpr int R0B = 2416, RLG = 2480, RLB = 2544, R1B = 2608, R2W = 2640, R2B = 3344;
constexpr int WC_N = 3368;

// ---------------------------------------------------------------------------
// Kernel 0: pack a row-major fp32 KxN weight matrix into bf16 B-fragments
// ---------------------------------------------------------------------------
__global__ __launch_bounds__(256) void pack_b(const float* __restrict__ W,
                                              unsigned int* __restrict__ dst,
                                              int K, int N) {
    int t = blockIdx.x * 256 + threadIdx.x;
    int total = (K * N) >> 1;
    if (t >= total) return;
    int frag = t >> 8;
    int lane = (t >> 3) & 31;
    int v    = t & 7;
    int kts  = K >> 5;
    int kt   = frag % kts;
    int nt   = frag / kts;
    int n    = nt * 16 + (lane & 15);
    int k0   = kt * 32 + (lane >> 4) * 16 + v * 2;
    dst[t] = pack2_bf16(W[k0 * N + n], W[(k0 + 1) * N + n]);
}

// ---------------------------------------------------------------------------
// WMMA operand helpers
// ---------------------------------------------------------------------------
static __device__ __forceinline__ v16bf load_bfrag(const unsigned int* __restrict__ base,
                                                   int frag, int lane) {
    const unsigned int* p = base + frag * 256 + lane * 8;
    Frag f;
    f.q[0] = ((const int4*)p)[0];
    f.q[1] = ((const int4*)p)[1];
    return f.v;
}

// Build A fragment (16x32 bf16) for k-tile kt from a fp32 row-major LDS matrix
// act[16][ld].  Layout: lane m (0..15) V0..7 hold K pairs 0..15;
// lanes 16..31 hold K 16..31 (contiguous, matches pack_b).
static __device__ __forceinline__ v16bf make_afrag(const float* __restrict__ act,
                                                   int ld, int kt, int lane) {
    int m = lane & 15, half = lane >> 4;
    const float* row = act + m * ld + kt * 32 + half * 16;
    Frag f;
#pragma unroll
    for (int v = 0; v < 8; ++v) {
        int koff = v * 2;
        f.u[v] = pack2_bf16(row[koff], row[koff + 1]);
    }
    return f.v;
}

// C/D layout: VGPR v, lane -> row = v + 8*(lane>>4), col = lane&15
static __device__ __forceinline__ void store_c_f32(float* __restrict__ dst, int ld,
                                                   int ncol0, v8f acc, int lane,
                                                   const float* __restrict__ bias,
                                                   bool relu) {
    int col = lane & 15, rbase = (lane >> 4) * 8;
    float b = bias[ncol0 + col];
#pragma unroll
    for (int v = 0; v < 8; ++v) {
        float x = acc[v] + b;
        if (relu) x = fmaxf(x, 0.f);
        dst[(rbase + v) * ld + ncol0 + col] = x;
    }
}

static __device__ __forceinline__ void store_c_bf(__bf16* __restrict__ dst, int ld,
                                                  int ncol0, v8f acc, int lane) {
    int col = lane & 15, rbase = (lane >> 4) * 8;
#pragma unroll
    for (int v = 0; v < 8; ++v)
        dst[(rbase + v) * ld + ncol0 + col] = (__bf16)acc[v];
}

// xs/hs = act(16x128) @ W(128x384), raw pre-activation, bf16 to LDS
static __device__ __forceinline__ void gemm_gru(const float* __restrict__ act,
                                                const unsigned int* __restrict__ wb,
                                                __bf16* __restrict__ dst,
                                                int wave, int lane) {
    v16bf a[4];
#pragma unroll
    for (int kt = 0; kt < 4; ++kt) a[kt] = make_afrag(act, 128, kt, lane);
    for (int nt = wave; nt < 24; nt += 8) {
        v8f acc = {};
#pragma unroll
        for (int kt = 0; kt < 4; ++kt) {
            v16bf b = load_bfrag(wb, nt * 4 + kt, lane);
            acc = __builtin_amdgcn_wmma_f32_16x16x32_bf16(false, a[kt], false, b,
                                                          (short)0, acc, false, false);
        }
        store_c_bf(dst, 384, nt * 16, acc, lane);
    }
}

// GRU gate math (keras reset_after=True), processed two lanes of n at a time.
// b layout (floats): [0..383]=input bias, [384..767]=recurrent bias.
static __device__ __forceinline__ void gru_gate(const __bf16* __restrict__ xs,
                                                const __bf16* __restrict__ hs,
                                                const float* __restrict__ b,
                                                float* __restrict__ h, int t) {
    const bf16x2* x2 = (const bf16x2*)xs;
    const bf16x2* g2 = (const bf16x2*)hs;
    const float2* b2 = (const float2*)b;
    float2*       hp = (float2*)h;
    for (int i = t; i < 1024; i += 256) {
        int r = i >> 6, n2 = i & 63;   // pair index: n = 2*n2
        int base = r * 192;            // 384 bf16 per row = 192 pairs
        bf16x2 xz = x2[base + n2],      hz = g2[base + n2];
        bf16x2 xg = x2[base + 64 + n2], hg = g2[base + 64 + n2];
        bf16x2 xh = x2[base + 128 + n2], hh = g2[base + 128 + n2];
        float2 bz = b2[n2],       uz = b2[192 + n2];
        float2 bg = b2[64 + n2],  ug = b2[256 + n2];
        float2 bh = b2[128 + n2], uh = b2[320 + n2];
        float2 hv = hp[r * 64 + n2];
        float2 res;
        {
            float z  = fsigmoid(((float)xz.x + bz.x) + ((float)hz.x + uz.x));
            float rr = fsigmoid(((float)xg.x + bg.x) + ((float)hg.x + ug.x));
            float cd = ftanh(((float)xh.x + bh.x) + rr * ((float)hh.x + uh.x));
            res.x = z * hv.x + (1.f - z) * cd;
        }
        {
            float z  = fsigmoid(((float)xz.y + bz.y) + ((float)hz.y + uz.y));
            float rr = fsigmoid(((float)xg.y + bg.y) + ((float)hg.y + ug.y));
            float cd = ftanh(((float)xh.y + bh.y) + rr * ((float)hh.y + uh.y));
            res.y = z * hv.y + (1.f - z) * cd;
        }
        hp[r * 64 + n2] = res;
    }
}

static __device__ __forceinline__ void ldcpy(float* dst, const float* __restrict__ src,
                                             int n, int t) {
    for (int i = t; i < n; i += 256) dst[i] = src[i];
}
// transpose a (nin x nout) row-major matrix into dst[k*nin + i] = src[i*nout + k]
static __device__ __forceinline__ void ldcpyT(float* dst, const float* __restrict__ src,
                                              int nin, int nout, int t) {
    for (int i = t; i < nin * nout; i += 256) {
        int k = i / nin, c = i % nin;
        dst[k * nin + c] = src[c * nout + k];
    }
}

static __device__ __forceinline__ float dot32(const float* __restrict__ a,
                                              const float* __restrict__ w) {
    const float4* a4 = (const float4*)a;
    const float4* w4 = (const float4*)w;
    float s = 0.f;
#pragma unroll
    for (int i = 0; i < 8; ++i) {
        float4 x = a4[i], y = w4[i];
        s += x.x * y.x + x.y * y.y + x.z * y.z + x.w * y.w;
    }
    return s;
}
static __device__ __forceinline__ float dot16(const float* __restrict__ a,
                                              const float* __restrict__ w) {
    const float4* a4 = (const float4*)a;
    const float4* w4 = (const float4*)w;
    float s = 0.f;
#pragma unroll
    for (int i = 0; i < 4; ++i) {
        float4 x = a4[i], y = w4[i];
        s += x.x * y.x + x.y * y.y + x.z * y.z + x.w * y.w;
    }
    return s;
}

// ---------------------------------------------------------------------------
// Kernel 1: per-sample conditioning (features, embeds, LN, cond/h1/h2, counts)
// ---------------------------------------------------------------------------
struct InitP {
    const float *ag0_W, *ag0_b, *ag_ln_g, *ag_ln_b, *ag1_W, *ag1_b;
    const float *ab0_W, *ab0_b, *ab_ln_g, *ab_ln_b, *ab1_W, *ab1_b;
    const float *init1, *init2;
};

static __device__ __forceinline__ float blk_sum128(float* red, int t, float v) {
    red[t] = v;
    __syncthreads();
    for (int s = 64; s > 0; s >>= 1) {
        if (t < s) red[t] += red[t + s];
        __syncthreads();
    }
    float r = red[0];
    __syncthreads();
    return r;
}

__global__ __launch_bounds__(128) void init_kernel(const float* __restrict__ ag,
                                                   const float* __restrict__ ab,
                                                   InitP P,
                                                   float* __restrict__ condg,
                                                   float* __restrict__ h1g,
                                                   float* __restrict__ h2g,
                                                   int* __restrict__ cnt) {
    __shared__ float feat_ag[68], feat_ab[68], hb[128], red[128], acc_cond[128];
    __shared__ int flags[92];
    int b = blockIdx.x, t = threadIdx.x;

    if (t < 34) {
        const float* sa = ag + (size_t)b * 320 * 34 + t;
        float s = 0.f, mx = -INFINITY;
        for (int r = 0; r < 320; ++r) { float v = sa[r * 34]; s += v; mx = fmaxf(mx, v); }
        feat_ag[t] = s / 320.f; feat_ag[34 + t] = mx;
        const float* sb = ab + (size_t)b * 460 * 34 + t;
        s = 0.f; mx = -INFINITY;
        for (int r = 0; r < 460; ++r) { float v = sb[r * 34]; s += v; mx = fmaxf(mx, v); }
        feat_ab[t] = s / 460.f; feat_ab[34 + t] = mx;
    }
    if (t < 92) {
        const float* pr = ab + (size_t)b * 460 * 34 + t * 170;
        int f = 0;
        for (int i = 0; i < 170; ++i) f |= (pr[i] != 0.f);
        flags[t] = f;
    }
    __syncthreads();
    if (t == 0) { int c = 0; for (int r = 0; r < 92; ++r) c += flags[r]; cnt[b] = c; }

    for (int which = 0; which < 2; ++which) {
        const float* feat = which ? feat_ab : feat_ag;
        const float* W0 = which ? P.ab0_W : P.ag0_W;
        const float* b0 = which ? P.ab0_b : P.ag0_b;
        const float* lg = which ? P.ab_ln_g : P.ag_ln_g;
        const float* lb = which ? P.ab_ln_b : P.ag_ln_b;
        const float* W1 = which ? P.ab1_W : P.ag1_W;
        const float* b1 = which ? P.ab1_b : P.ag1_b;

        float s = b0[t];
        for (int i = 0; i < 68; ++i) s += feat[i] * W0[i * 128 + t];
        s = fmaxf(s, 0.f);
        float mean = blk_sum128(red, t, s) * (1.f / 128.f);
        float d = s - mean;
        float var = blk_sum128(red, t, d * d) * (1.f / 128.f);
        float nrm = d * rsqrtf(var + 1e-3f) * lg[t] + lb[t];
        hb[t] = nrm;
        __syncthreads();
        float s2 = b1[t];
        for (int i = 0; i < 128; ++i) s2 += hb[i] * W1[i * 128 + t];
        s2 = fmaxf(s2, 0.f);
        if (which == 0) acc_cond[t] = s2; else acc_cond[t] += s2;
        __syncthreads();
    }
    float c = acc_cond[t];
    condg[(size_t)b * 128 + t] = c;
    h1g[(size_t)b * 128 + t] = P.init1[t] + c;
    h2g[(size_t)b * 128 + t] = P.init2[t] + c;
}

// ---------------------------------------------------------------------------
// Kernel 2: 460-step autoregressive scan.  64 WGs x 256 threads (8 wave32),
// each WG owns a 16-row batch tile; all GEMMs via v_wmma_f32_16x16x32_bf16.
// ---------------------------------------------------------------------------
struct ScanP {
    const float *gru1_b, *gru2_b;
    const float *c0_b, *c_ln_g, *c_ln_b, *c1_b, *c2_W, *c2_b;
    const float *q0_b, *q1_b, *q2_W, *q2_b;
    const float *e0_b, *e_ln_g, *e_ln_b, *e1_b, *e2_W, *e2_b;
    const float *r0_b, *r_ln_g, *r_ln_b, *r1_b, *r2_W, *r2_b;
};

__global__ __launch_bounds__(256) void scan_kernel(const unsigned int* __restrict__ wp,
                                                   const float* __restrict__ condg,
                                                   const float* __restrict__ h1g,
                                                   const float* __restrict__ h2g,
                                                   const int* __restrict__ cnt,
                                                   ScanP P,
                                                   float* __restrict__ out) {
    __shared__ __align__(16) char smem[62776];
    float*  s_cond = (float*)(smem + 0);          // 16x128 f32
    float*  s_h1   = (float*)(smem + 8192);       // 16x128 f32
    float*  s_h2   = (float*)(smem + 16384);      // 16x128 f32
    __bf16* s_xsb  = (__bf16*)(smem + 24576);     // 16x384 bf16
    __bf16* s_hsb  = (__bf16*)(smem + 36864);     // 16x384 bf16
    // head buffers alias the xs/hs region (dead during head phases)
    float* s_hc64  = (float*)(smem + 24576);
    float* s_he64  = (float*)(smem + 28672);
    float* s_hr64  = (float*)(smem + 32768);
    float* s_hq32  = (float*)(smem + 36864);
    float* s_hc32  = (float*)(smem + 38912);
    float* s_he32  = (float*)(smem + 40960);
    float* s_hr32  = (float*)(smem + 43008);
    float* s_hq16  = (float*)(smem + 45056);
    float* s_logit = (float*)(smem + 46080);      // 16x34 f32
    float* wc      = (float*)(smem + 49152);      // const table (3368 f32)
    float* smean   = (float*)(smem + 62624);      // 16
    int*   scon    = (int*)  (smem + 62688);      // 16

    int t = threadIdx.x;
    int wave = t >> 5, lane = t & 31;
    int row0 = blockIdx.x * 16;

    // load recurrent state tile
    for (int i = t; i < 2048; i += 256) {
        int r = i >> 7, c = i & 127;
        s_cond[i] = condg[(size_t)(row0 + r) * 128 + c];
        s_h1[i]   = h1g[(size_t)(row0 + r) * 128 + c];
        s_h2[i]   = h2g[(size_t)(row0 + r) * 128 + c];
    }
    if (t < 16) scon[t] = cnt[row0 + t];

    // preload biases / LN params / tiny final weights (transposed) into LDS
    ldcpy(wc + G1B, P.gru1_b, 768, t);  ldcpy(wc + G2B, P.gru2_b, 768, t);
    ldcpy(wc + C0B, P.c0_b, 64, t);     ldcpy(wc + CLG, P.c_ln_g, 64, t);
    ldcpy(wc + CLB, P.c_ln_b, 64, t);   ldcpy(wc + C1B, P.c1_b, 32, t);
    ldcpyT(wc + C2W, P.c2_W, 32, 3, t); ldcpy(wc + C2B, P.c2_b, 3, t);
    ldcpy(wc + Q0B, P.q0_b, 32, t);     ldcpy(wc + Q1B, P.q1_b, 16, t);
    ldcpy(wc + Q2W, P.q2_W, 16, t);     ldcpy(wc + Q2B, P.q2_b, 1, t);
    ldcpy(wc + E0B, P.e0_b, 64, t);     ldcpy(wc + ELG, P.e_ln_g, 64, t);
    ldcpy(wc + ELB, P.e_ln_b, 64, t);   ldcpy(wc + E1B, P.e1_b, 32, t);
    ldcpyT(wc + E2W, P.e2_W, 32, 8, t); ldcpy(wc + E2B, P.e2_b, 8, t);
    ldcpy(wc + R0B, P.r0_b, 64, t);     ldcpy(wc + RLG, P.r_ln_g, 64, t);
    ldcpy(wc + RLB, P.r_ln_b, 64, t);   ldcpy(wc + R1B, P.r1_b, 32, t);
    ldcpyT(wc + R2W, P.r2_W, 32, 22, t); ldcpy(wc + R2B, P.r2_b, 22, t);
    __syncthreads();

#pragma unroll 1
    for (int step = 0; step < 460; ++step) {
        // --- GRU1: xs = cond@W1, hs = h1@U1 (WMMA) ---
        gemm_gru(s_cond, wp + OFF_G1W, s_xsb, wave, lane);
        gemm_gru(s_h1,   wp + OFF_G1U, s_hsb, wave, lane);
        __syncthreads();
        gru_gate(s_xsb, s_hsb, wc + G1B, s_h1, t);   // h1 <- new h1 (== o1)
        __syncthreads();
        // --- GRU2: xs = o1@W2, hs = h2@U2 ---
        gemm_gru(s_h1, wp + OFF_G2W, s_xsb, wave, lane);
        gemm_gru(s_h2, wp + OFF_G2U, s_hsb, wave, lane);
        __syncthreads();
        gru_gate(s_xsb, s_hsb, wc + G2B, s_h2, t);   // h2 <- new h2 (== o2)
        __syncthreads();

        // --- head stage 1: o2 @ {c0,e0,r0,q0}, bias+relu (14 WMMA tile jobs) ---
        {
            v16bf a[4];
#pragma unroll
            for (int kt = 0; kt < 4; ++kt) a[kt] = make_afrag(s_h2, 128, kt, lane);
            for (int j = wave; j < 14; j += 8) {
                const unsigned int* wb; float* dst; const float* bias; int nt, ld;
                if (j < 4)       { wb = wp + OFF_C0; dst = s_hc64; bias = wc + C0B; nt = j;      ld = 64; }
                else if (j < 8)  { wb = wp + OFF_E0; dst = s_he64; bias = wc + E0B; nt = j - 4;  ld = 64; }
                else if (j < 12) { wb = wp + OFF_R0; dst = s_hr64; bias = wc + R0B; nt = j - 8;  ld = 64; }
                else             { wb = wp + OFF_Q0; dst = s_hq32; bias = wc + Q0B; nt = j - 12; ld = 32; }
                v8f acc = {};
#pragma unroll
                for (int kt = 0; kt < 4; ++kt) {
                    v16bf b = load_bfrag(wb, nt * 4 + kt, lane);
                    acc = __builtin_amdgcn_wmma_f32_16x16x32_bf16(false, a[kt], false, b,
                                                                  (short)0, acc, false, false);
                }
                store_c_f32(dst, ld, nt * 16, acc, lane, bias, true);
            }
        }
        __syncthreads();

        // --- LayerNorm(64) for c/e/r heads: 48 (row,head) pairs, float4 ---
        if (t < 48) {
            int r = t / 3, hd = t % 3;
            float* buf = hd == 0 ? s_hc64 : (hd == 1 ? s_he64 : s_hr64);
            const float4* g4 = (const float4*)(wc + (hd == 0 ? CLG : (hd == 1 ? ELG : RLG)));
            const float4* b4 = (const float4*)(wc + (hd == 0 ? CLB : (hd == 1 ? ELB : RLB)));
            float4* p4 = (float4*)(buf + r * 64);
            float mu = 0.f;
#pragma unroll
            for (int i = 0; i < 16; ++i) { float4 x = p4[i]; mu += x.x + x.y + x.z + x.w; }
            mu *= (1.f / 64.f);
            float vr = 0.f;
#pragma unroll
            for (int i = 0; i < 16; ++i) {
                float4 x = p4[i];
                float a = x.x - mu, b = x.y - mu, c = x.z - mu, d = x.w - mu;
                vr += a * a + b * b + c * c + d * d;
            }
            vr *= (1.f / 64.f);
            float inv = rsqrtf(vr + 1e-3f);
#pragma unroll
            for (int i = 0; i < 16; ++i) {
                float4 x = p4[i], g = g4[i], bb = b4[i];
                x.x = (x.x - mu) * inv * g.x + bb.x;
                x.y = (x.y - mu) * inv * g.y + bb.y;
                x.z = (x.z - mu) * inv * g.z + bb.z;
                x.w = (x.w - mu) * inv * g.w + bb.w;
                p4[i] = x;
            }
        }
        __syncthreads();

        // --- head stage 2: {c1,e1,r1,q1} WMMA (7 tile jobs) ---
        if (wave < 7) {
            int j = wave;
            const float* src; const unsigned int* wb; float* dst; const float* bias;
            int nt, kts, lds, ldd;
            if (j < 2)      { src = s_hc64; wb = wp + OFF_C1; dst = s_hc32; bias = wc + C1B; nt = j;     kts = 2; lds = 64; ldd = 32; }
            else if (j < 4) { src = s_he64; wb = wp + OFF_E1; dst = s_he32; bias = wc + E1B; nt = j - 2; kts = 2; lds = 64; ldd = 32; }
            else if (j < 6) { src = s_hr64; wb = wp + OFF_R1; dst = s_hr32; bias = wc + R1B; nt = j - 4; kts = 2; lds = 64; ldd = 32; }
            else            { src = s_hq32; wb = wp + OFF_Q1; dst = s_hq16; bias = wc + Q1B; nt = 0;     kts = 1; lds = 32; ldd = 16; }
            v8f acc = {};
            for (int kt = 0; kt < kts; ++kt) {
                v16bf a = make_afrag(src, lds, kt, lane);
                v16bf b = load_bfrag(wb, nt * kts + kt, lane);
                acc = __builtin_amdgcn_wmma_f32_16x16x32_bf16(false, a, false, b,
                                                              (short)0, acc, false, false);
            }
            store_c_f32(dst, ldd, nt * 16, acc, lane, bias, true);
        }
        __syncthreads();

        // --- all 544 final logits in parallel (k-major: <=2 branch paths/wave) ---
        for (int j = t; j < 544; j += 256) {
            int k = j >> 4, r = j & 15;
            float v;
            if (k < 3) {
                v = dot32(s_hc32 + r * 32, wc + C2W + k * 32) + wc[C2B + k];
            } else if (k == 3) {
                v = ftanh(dot16(s_hq16 + r * 16, wc + Q2W) + wc[Q2B]);
            } else if (k < 12) {
                int e = k - 4;
                v = dot32(s_he32 + r * 32, wc + E2W + e * 32) + wc[E2B + e];
            } else {
                int rr = k - 12;
                v = dot32(s_hr32 + r * 32, wc + R2W + rr * 32) + wc[R2B + rr];
            }
            s_logit[r * 34 + k] = v;
        }
        __syncthreads();

        // --- argmax / one-hot / masked output / cond feedback scalar ---
        if (t < 16) {
            int r = t;
            const float* lg = s_logit + r * 34;
            float atom[34];
            atom[0] = lg[0]; atom[1] = lg[1]; atom[2] = lg[2]; atom[3] = lg[3];
            int be = 0; float bv = lg[4];
            for (int k = 1; k < 8; ++k) { float s = lg[4 + k]; if (s > bv) { bv = s; be = k; } }
            for (int k = 0; k < 8; ++k) atom[4 + k] = (k == be) ? 1.f : 0.f;
            int br = 0; bv = lg[12];
            for (int k = 1; k < 22; ++k) { float s = lg[12 + k]; if (s > bv) { bv = s; br = k; } }
            for (int k = 0; k < 22; ++k) atom[12 + k] = (k == br) ? 1.f : 0.f;

            float m = ((step / 5) < scon[r]) ? 1.f : 0.f;
            float ssum = 0.f;
            float* og = out + (size_t)(row0 + r) * 15640 + step * 34;
            for (int k = 0; k < 34; ++k) { float v = atom[k] * m; og[k] = v; ssum += v; }
            smean[r] = ssum * (0.1f / 34.f);
        }
        __syncthreads();
        for (int i = t; i < 2048; i += 256) s_cond[i] += smean[i >> 7];
        __syncthreads();
    }
}

// ---------------------------------------------------------------------------
// Host glue.  Input order = setup_inputs() dict order:
// d_in[0]=ag_input, d_in[1]=ab_input, d_in[2..51]=params in insertion order.
// ---------------------------------------------------------------------------
extern "C" void kernel_launch(void* const* d_in, const int* in_sizes, int n_in,
                              void* d_out, int out_size, void* d_ws, size_t ws_size,
                              hipStream_t stream) {
    (void)in_sizes; (void)n_in; (void)out_size; (void)ws_size;
    auto F = [&](int i) { return (const float*)d_in[i]; };
    const float* ag_input = F(0);
    const float* ab_input = F(1);
    unsigned int* wp  = (unsigned int*)d_ws;
    float*        wsf = (float*)d_ws;

    auto packLaunch = [&](const float* W, int off, int K, int N) {
        int total = (K * N) / 2;
        pack_b<<<dim3((total + 255) / 256), dim3(256), 0, stream>>>(W, wp + off, K, N);
    };
    packLaunch(F(14), OFF_G1W, 128, 384);
    packLaunch(F(15), OFF_G1U, 128, 384);
    packLaunch(F(17), OFF_G2W, 128, 384);
    packLaunch(F(18), OFF_G2U, 128, 384);
    packLaunch(F(20), OFF_C0, 128, 64);
    packLaunch(F(34), OFF_E0, 128, 64);
    packLaunch(F(42), OFF_R0, 128, 64);
    packLaunch(F(28), OFF_Q0, 128, 32);
    packLaunch(F(24), OFF_C1, 64, 32);
    packLaunch(F(38), OFF_E1, 64, 32);
    packLaunch(F(46), OFF_R1, 64, 32);
    packLaunch(F(30), OFF_Q1, 32, 16);

    InitP ip = { F(2), F(3), F(4), F(5), F(6), F(7),
                 F(8), F(9), F(10), F(11), F(12), F(13),
                 F(50), F(51) };
    init_kernel<<<dim3(1024), dim3(128), 0, stream>>>(
        ag_input, ab_input, ip,
        wsf + OFF_CONDF, wsf + OFF_H1F, wsf + OFF_H2F, (int*)(wp + OFF_CNT));

    ScanP sp = { F(16), F(19),
                 F(21), F(22), F(23), F(25), F(26), F(27),
                 F(29), F(31), F(32), F(33),
                 F(35), F(36), F(37), F(39), F(40), F(41),
                 F(43), F(44), F(45), F(47), F(48), F(49) };
    scan_kernel<<<dim3(64), dim3(256), 0, stream>>>(
        wp, wsf + OFF_CONDF, wsf + OFF_H1F, wsf + OFF_H2F,
        (const int*)(wp + OFF_CNT), sp, (float*)d_out);
}
